// GraphRWKV_GNN_Model_70265664962635
// MI455X (gfx1250) — matrix operand order repaired and checked by reading, not compiled, not run on hardware
//
#include <hip/hip_runtime.h>
#include <hip/hip_bf16.h>
#include <math.h>

// ---------------------------------------------------------------- types
typedef __attribute__((ext_vector_type(16))) __bf16       v16bf;
typedef __attribute__((ext_vector_type(8)))  float        v8f;
typedef __attribute__((ext_vector_type(8)))  unsigned int v8u;

union ABits { v8u u; v16bf b; };

#define NN     4096
#define DEMB   512
#define DGNN   256
#define NHEAD  2
#define DHEAD  128
#define FUS    768
#define FHID   384
#define JSPLIT 8
#define JCHUNK (NN / JSPLIT)

// ---------------------------------------------------------------- helpers
__device__ __forceinline__ unsigned short f2bf(float x) {
  unsigned int u = __float_as_uint(x);
  unsigned int r = (u + 0x7FFFu + ((u >> 16) & 1u)) >> 16;
  return (unsigned short)r;
}
__device__ __forceinline__ float bf2f(unsigned short h) {
  return __uint_as_float(((unsigned int)h) << 16);
}
__device__ __forceinline__ float gelu_exact(float x) {
  return 0.5f * x * (1.0f + erff(x * 0.70710678118654752f));
}
// wave-uniform value -> SGPR (scalar loop control, scalar addressing)
__device__ __forceinline__ int uniform(int x) {
  return __builtin_amdgcn_readfirstlane(x);
}

// A tile 16x32 bf16, row-major source (lda = K). CDNA5 A layout:
// lane L: m = L&15 ; lanes<16 kbase=0, lanes>=16 kbase=8
// vgpr v<4: k = kbase+2v ; v>=4: k = 16+kbase+2(v-4)   (pairs contiguous in K)
__device__ __forceinline__ v16bf load_A_tile(const unsigned short* __restrict__ A,
                                             int lda, int m0, int k0) {
  int L = threadIdx.x & 31;
  int m = m0 + (L & 15);
  int kb = (L < 16) ? 0 : 8;
  ABits r;
#pragma unroll
  for (int v = 0; v < 8; ++v) {
    int k = (v < 4) ? (kb + 2 * v) : (16 + kb + 2 * (v - 4));
    r.u[v] = *(const unsigned int*)(A + (size_t)m * lda + k0 + k);
  }
  return r.b;
}

// B tile 32x16 bf16 from BT (N x K row-major, ldbt = K). CDNA5 B layout:
// lane L: n = L&15 ; same K order as A: kbase=0/8, v<4: k=kbase+2v ; v>=4: k=16+kbase+2(v-4)
__device__ __forceinline__ v16bf load_Bt_tile(const unsigned short* __restrict__ BT,
                                              int ldbt, int n0, int k0) {
  int L = threadIdx.x & 31;
  int n = n0 + (L & 15);
  int kb = (L < 16) ? 0 : 8;
  ABits r;
#pragma unroll
  for (int v = 0; v < 8; ++v) {
    int k = (v < 4) ? (kb + 2 * v) : (16 + kb + 2 * (v - 4));
    r.u[v] = *(const unsigned int*)(BT + (size_t)n * ldbt + k0 + k);
  }
  return r.b;
}

// ---------------------------------------------------------------- converts
__global__ void k_f32_to_bf16(const float* __restrict__ in,
                              unsigned short* __restrict__ out, int n) {
  int t = blockIdx.x * blockDim.x + threadIdx.x;
  if (t < n) out[t] = f2bf(in[t]);
}

// in: R x C fp32 row-major  ->  out: C x R bf16 row-major (i.e. B^T)
__global__ void k_transpose_bf16(const float* __restrict__ in,
                                 unsigned short* __restrict__ out, int R, int C) {
  int t = blockIdx.x * blockDim.x + threadIdx.x;
  if (t < R * C) {
    int r = t / C, c = t % C;
    out[(size_t)c * R + r] = f2bf(in[t]);
  }
}

// ---------------------------------------------------------------- WMMA GEMM
// C(MxN) = A(MxK) @ B(KxN), A bf16 row-major, B supplied transposed (NxK bf16).
// Each wave computes a 16x64 strip: stage A + 4 B tiles (one load clause),
// then issue 4 WMMA back-to-back.
// MODE 0: fp32 out row-major + bias
// MODE 1: bf16 out TRANSPOSED (CT is N x M), no bias
// MODE 2: fp32 out row-major + bias + exact GELU
template <int MODE>
__global__ void k_gemm_bf16(const unsigned short* __restrict__ A,
                            const unsigned short* __restrict__ BT,
                            void* __restrict__ Cout,
                            const float* __restrict__ bias,
                            int M, int N, int K) {
  int wave = uniform(blockIdx.x * (blockDim.x >> 5) + (threadIdx.x >> 5));
  int strips = N >> 6;                 // 64-wide output strips
  int m0 = (wave / strips) << 4;
  int n0 = (wave % strips) << 6;
  v8f acc[4] = {};
  for (int k0 = 0; k0 < K; k0 += 32) {
    v16bf a = load_A_tile(A, K, m0, k0);
    v16bf bt[4];
#pragma unroll
    for (int t = 0; t < 4; ++t) bt[t] = load_Bt_tile(BT, K, n0 + 16 * t, k0);
#pragma unroll
    for (int t = 0; t < 4; ++t)
      acc[t] = __builtin_amdgcn_wmma_f32_16x16x32_bf16(false, a, false, bt[t],
                                                       (short)0, acc[t], false, false);
  }
  int L = threadIdx.x & 31;
  int mb = (L < 16) ? 0 : 8;
#pragma unroll
  for (int t = 0; t < 4; ++t) {
    int n = n0 + 16 * t + (L & 15);
    if (MODE == 1) {
      unsigned short* CT = (unsigned short*)Cout;
      unsigned int* p = (unsigned int*)(CT + (size_t)n * M + m0 + mb);
#pragma unroll
      for (int r = 0; r < 4; ++r)
        p[r] = (unsigned int)f2bf(acc[t][2 * r]) |
               ((unsigned int)f2bf(acc[t][2 * r + 1]) << 16);
    } else {
      float* C = (float*)Cout;
      float bv = bias ? bias[n] : 0.0f;
#pragma unroll
      for (int r = 0; r < 8; ++r) {
        float x = acc[t][r] + bv;
        if (MODE == 2) x = gelu_exact(x);
        C[(size_t)(m0 + mb + r) * N + n] = x;
      }
    }
  }
}

// ---------------------------------------------------------------- scatter-mean
__global__ void k_scatter_add(const float* __restrict__ hg, const int* __restrict__ idx,
                              float* __restrict__ sums, float* __restrict__ counts) {
  int b = blockIdx.x, c = threadIdx.x;
  int n = idx[b];
  atomicAdd(&sums[(size_t)n * DGNN + c], hg[(size_t)b * DGNN + c]);
  if (c == 0) atomicAdd(&counts[n], 1.0f);
}

__global__ void k_select_subfeats(const float* __restrict__ sums,
                                  const float* __restrict__ counts,
                                  const float* __restrict__ node_emb,
                                  unsigned short* __restrict__ subf_bf) {
  int t = blockIdx.x * blockDim.x + threadIdx.x;
  int n = t / DGNN;
  float cnt = counts[n];
  float v = (cnt > 0.0f) ? (sums[t] / cnt) : node_emb[t];
  subf_bf[t] = f2bf(v);
}

// ---------------------------------------------------------------- attention coeffs
__global__ void k_attn_coef(const unsigned short* __restrict__ WhT,
                            const float* __restrict__ a_gat,
                            float* __restrict__ f_src, float* __restrict__ f_dst) {
  int t = blockIdx.x * blockDim.x + threadIdx.x;  // 0..2*NN-1
  int h = t >> 12;
  int n = t & (NN - 1);
  const unsigned short* base = WhT + (size_t)h * DHEAD * NN + n;
  const float* a1 = a_gat + h * (2 * DHEAD);
  float fs = 0.f, fd = 0.f;
#pragma unroll 4
  for (int d = 0; d < DHEAD; ++d) {
    float w = bf2f(base[(size_t)d * NN]);
    fs += w * a1[d];
    fd += w * a1[DHEAD + d];
  }
  f_src[t] = fs;
  f_dst[t] = fd;
}

// ---------------------------------------------------------------- softmax stats
// one wave per (head, row): streaming online max/sum, branchless mask.
__global__ void k_gat_softmax_stats(const float* __restrict__ adj,
                                    const float* __restrict__ f_src,
                                    const float* __restrict__ f_dst,
                                    float* __restrict__ rmax, float* __restrict__ rsum) {
  int wave = uniform(blockIdx.x * (blockDim.x >> 5) + (threadIdx.x >> 5));
  int h = wave >> 12;
  int i = wave & (NN - 1);
  int L = threadIdx.x & 31;
  float fs = f_src[h * NN + i];
  const float* adjrow = adj + (size_t)i * NN;
  const float* fdh = f_dst + h * NN;
  float m = -3.0e38f, l = 0.0f;
  for (int j = L; j < NN; j += 32) {
    float ad = adjrow[j];
    float e = fs + fdh[j];
    e = (e > 0.0f) ? e : 0.2f * e;           // leaky relu, alpha=0.2
    e = (ad > 0.0f) ? e : -3.0e38f;          // mask (cndmask, no branch)
    float mn = fmaxf(m, e);
    float t = __expf(e - mn);                // unconditional exp
    t = (ad > 0.0f) ? t : 0.0f;
    l = l * __expf(m - mn) + t;
    m = mn;
  }
#pragma unroll
  for (int off = 16; off; off >>= 1) {
    float m2 = __shfl_xor(m, off, 32);
    float l2 = __shfl_xor(l, off, 32);
    float mn = fmaxf(m, m2);
    l = l * __expf(m - mn) + l2 * __expf(m2 - mn);
    m = mn;
  }
  if (L == 0) { rmax[h * NN + i] = m; rsum[h * NN + i] = l; }
}

// ---------------------------------------------------------------- attention aggregate
// wave = (head, 16-row block, j-chunk) -- all scalar via readfirstlane.
// Softmax probs built directly in WMMA A-layout registers (branchless), all 8
// B tiles staged first (single load clause + one wait), then 8 WMMA issued
// back-to-back. Partial 16x128 fp32 tiles per j-chunk; chunk reduction is
// folded into the ELU+LN kernel.
__global__ void k_gat_aggregate(const float* __restrict__ adj,
                                const float* __restrict__ f_src,
                                const float* __restrict__ f_dst,
                                const float* __restrict__ rmax,
                                const float* __restrict__ rsum,
                                const unsigned short* __restrict__ WhT,
                                float* __restrict__ hp_part) {
  int wave = uniform(blockIdx.x * (blockDim.x >> 5) + (threadIdx.x >> 5));
  int c8 = wave & (JSPLIT - 1);
  int ib = (wave >> 3) & 255;
  int h = wave >> 11;
  int i0 = ib << 4;
  int L = threadIdx.x & 31;
  int m = L & 15;
  int i = i0 + m;
  int kb = (L < 16) ? 0 : 8;
  float fs = f_src[h * NN + i];
  float rm = rmax[h * NN + i];
  float inv = 1.0f / rsum[h * NN + i];
  const float* adjrow = adj + (size_t)i * NN;
  const float* fdh = f_dst + h * NN;
  const unsigned short* Bt = WhT + (size_t)h * DHEAD * NN;
  v8f acc[8] = {};
  int jbeg = c8 * JCHUNK;
  for (int j0 = jbeg; j0 < jbeg + JCHUNK; j0 += 32) {
    __builtin_prefetch(adjrow + j0 + 256, 0, 1);
    // stage all 8 B tiles: one big load clause, one wait
    v16bf bt[8];
#pragma unroll
    for (int d = 0; d < 8; ++d) bt[d] = load_Bt_tile(Bt, NN, d * 16, j0);
    // build softmax-prob A tile in WMMA layout (branchless)
    ABits a;
#pragma unroll
    for (int v = 0; v < 8; ++v) {
      int k = (v < 4) ? (kb + 2 * v) : (16 + kb + 2 * (v - 4));
      int j = j0 + k;
      float ad0 = adjrow[j], ad1 = adjrow[j + 1];
      float e0 = fs + fdh[j];     e0 = (e0 > 0.0f) ? e0 : 0.2f * e0;
      float e1 = fs + fdh[j + 1]; e1 = (e1 > 0.0f) ? e1 : 0.2f * e1;
      float w0 = (ad0 > 0.0f) ? inv : 0.0f;   // mask via multiplier: no branch
      float w1 = (ad1 > 0.0f) ? inv : 0.0f;
      float p0 = __expf(e0 - rm) * w0;
      float p1 = __expf(e1 - rm) * w1;
      a.u[v] = (unsigned int)f2bf(p0) | ((unsigned int)f2bf(p1) << 16);
    }
#pragma unroll
    for (int d = 0; d < 8; ++d)
      acc[d] = __builtin_amdgcn_wmma_f32_16x16x32_bf16(false, a.b, false, bt[d],
                                                       (short)0, acc[d], false, false);
  }
  float* dst = hp_part + (size_t)c8 * NN * DGNN;
  int nloc = L & 15;
  int mb = (L < 16) ? 0 : 8;
#pragma unroll
  for (int d = 0; d < 8; ++d)
#pragma unroll
    for (int r = 0; r < 8; ++r)
      dst[(size_t)(i0 + mb + r) * DGNN + h * DHEAD + d * 16 + nloc] = acc[d][r];
}

// ---------------------------------------------- j-chunk reduce + ELU + LayerNorm (256)
__global__ void k_elu_ln(const float* __restrict__ hp_part, const float* __restrict__ g,
                         const float* __restrict__ b, float* __restrict__ out) {
  __shared__ float s1[256], s2[256];
  int n = blockIdx.x, c = threadIdx.x;
  float x = 0.0f;
#pragma unroll
  for (int c8 = 0; c8 < JSPLIT; ++c8)
    x += hp_part[(size_t)c8 * NN * DGNN + (size_t)n * DGNN + c];
  x = (x > 0.0f) ? x : (__expf(x) - 1.0f);
  s1[c] = x; s2[c] = x * x;
  __syncthreads();
  for (int off = 128; off; off >>= 1) {
    if (c < off) { s1[c] += s1[c + off]; s2[c] += s2[c + off]; }
    __syncthreads();
  }
  float mean = s1[0] * (1.0f / DGNN);
  float var = s2[0] * (1.0f / DGNN) - mean * mean;
  out[(size_t)n * DGNN + c] = (x - mean) * rsqrtf(var + 1e-5f) * g[c] + b[c];
}

// ------------------------------------------------- gather + concat + LayerNorm (768) -> bf16
__global__ void k_fuse_ln(const float* __restrict__ hT, const float* __restrict__ sub_out,
                          const int* __restrict__ idx, const float* __restrict__ g2,
                          const float* __restrict__ bb2, unsigned short* __restrict__ z_bf) {
  __shared__ float s1[256], s2[256];
  int bb = blockIdx.x, c = threadIdx.x;
  int n = idx[bb];
  float v[3], a1 = 0.f, a2 = 0.f;
#pragma unroll
  for (int r = 0; r < 3; ++r) {
    int f = c + r * 256;
    float x = (f < DEMB) ? hT[(size_t)bb * DEMB + f]
                         : sub_out[(size_t)n * DGNN + (f - DEMB)];
    v[r] = x; a1 += x; a2 += x * x;
  }
  s1[c] = a1; s2[c] = a2;
  __syncthreads();
  for (int off = 128; off; off >>= 1) {
    if (c < off) { s1[c] += s1[c + off]; s2[c] += s2[c + off]; }
    __syncthreads();
  }
  float mean = s1[0] * (1.0f / FUS);
  float var = s2[0] * (1.0f / FUS) - mean * mean;
  float rs = rsqrtf(var + 1e-5f);
#pragma unroll
  for (int r = 0; r < 3; ++r) {
    int f = c + r * 256;
    z_bf[(size_t)bb * FUS + f] = f2bf((v[r] - mean) * rs * g2[f] + bb2[f]);
  }
}

// ---------------------------------------------------------------- final 384 -> 1
__global__ void k_head(const float* __restrict__ z2, const float* __restrict__ W2,
                       const float* __restrict__ b2, float* __restrict__ out) {
  int wave = uniform(blockIdx.x * (blockDim.x >> 5) + (threadIdx.x >> 5));
  int L = threadIdx.x & 31;
  const float* row = z2 + (size_t)wave * FHID;
  float acc = 0.0f;
  for (int k = L; k < FHID; k += 32) acc += row[k] * W2[k];
#pragma unroll
  for (int off = 16; off; off >>= 1) acc += __shfl_xor(acc, off, 32);
  if (L == 0) out[wave] = acc + b2[0];
}

// ---------------------------------------------------------------- launcher
extern "C" void kernel_launch(void* const* d_in, const int* in_sizes, int n_in,
                              void* d_out, int out_size, void* d_ws, size_t ws_size,
                              hipStream_t stream) {
  const float* h_temporal = (const float*)d_in[0];
  const int*   node_idx   = (const int*)d_in[1];
  const float* adj        = (const float*)d_in[2];
  const float* W_proj     = (const float*)d_in[3];
  const float* b_proj     = (const float*)d_in[4];
  const float* W_gat      = (const float*)d_in[5];
  const float* a_gat      = (const float*)d_in[6];
  const float* node_emb   = (const float*)d_in[7];
  const float* ln_g       = (const float*)d_in[8];
  const float* ln_b       = (const float*)d_in[9];
  const float* ln2_g      = (const float*)d_in[10];
  const float* ln2_b      = (const float*)d_in[11];
  const float* W1         = (const float*)d_in[12];
  const float* b1         = (const float*)d_in[13];
  const float* W2         = (const float*)d_in[14];
  const float* b2         = (const float*)d_in[15];
  float* out = (float*)d_out;

  // workspace bump allocator (256B aligned)
  char* base = (char*)d_ws;
  size_t off = 0;
  auto alloc = [&](size_t bytes) -> void* {
    off = (off + 255) & ~(size_t)255;
    void* p = base + off;
    off += bytes;
    return p;
  };
  unsigned short* hT_bf    = (unsigned short*)alloc((size_t)NN * DEMB * 2);
  unsigned short* WprojT   = (unsigned short*)alloc((size_t)DGNN * DEMB * 2);
  unsigned short* WgatT    = (unsigned short*)alloc((size_t)DGNN * DGNN * 2);
  unsigned short* W1T      = (unsigned short*)alloc((size_t)FHID * FUS * 2);
  float*          h_gnn_in = (float*)alloc((size_t)NN * DGNN * 4);
  float*          sums     = (float*)alloc((size_t)NN * DGNN * 4);
  float*          counts   = (float*)alloc((size_t)NN * 4);
  unsigned short* subf_bf  = (unsigned short*)alloc((size_t)NN * DGNN * 2);
  unsigned short* WhT      = (unsigned short*)alloc((size_t)DGNN * NN * 2);
  float*          f_src    = (float*)alloc((size_t)NHEAD * NN * 4);
  float*          f_dst    = (float*)alloc((size_t)NHEAD * NN * 4);
  float*          rmax     = (float*)alloc((size_t)NHEAD * NN * 4);
  float*          rsum     = (float*)alloc((size_t)NHEAD * NN * 4);
  float*          hp_part  = (float*)alloc((size_t)JSPLIT * NN * DGNN * 4);
  float*          sub_out  = (float*)alloc((size_t)NN * DGNN * 4);
  unsigned short* z_bf     = (unsigned short*)alloc((size_t)NN * FUS * 2);
  float*          z2       = (float*)alloc((size_t)NN * FHID * 4);
  (void)ws_size; (void)n_in; (void)in_sizes; (void)out_size;

  // 0) zero scatter buffers (re-done every call: deterministic)
  hipMemsetAsync(sums, 0, (size_t)NN * DGNN * 4, stream);
  hipMemsetAsync(counts, 0, (size_t)NN * 4, stream);

  // 1) bf16 conversions / weight transposes
  k_f32_to_bf16<<<(NN * DEMB + 255) / 256, 256, 0, stream>>>(h_temporal, hT_bf, NN * DEMB);
  k_transpose_bf16<<<(DEMB * DGNN + 255) / 256, 256, 0, stream>>>(W_proj, WprojT, DEMB, DGNN);
  // W_gat: (H,256,128) -> BT rows hd = h*128+d of length 256
  k_transpose_bf16<<<(DGNN * DHEAD + 255) / 256, 256, 0, stream>>>(
      W_gat, WgatT, DGNN, DHEAD);
  k_transpose_bf16<<<(DGNN * DHEAD + 255) / 256, 256, 0, stream>>>(
      W_gat + (size_t)DGNN * DHEAD, WgatT + (size_t)DHEAD * DGNN, DGNN, DHEAD);
  k_transpose_bf16<<<(FUS * FHID + 255) / 256, 256, 0, stream>>>(W1, W1T, FUS, FHID);

  // 2) GEMM1: h_gnn_in = h_temporal @ W_proj + b_proj   (4096x512x256)
  k_gemm_bf16<0><<<(NN / 16) * (DGNN / 64) / 4, 128, 0, stream>>>(
      hT_bf, WprojT, h_gnn_in, b_proj, NN, DGNN, DEMB);

  // 3) scatter-mean onto nodes, select vs learned embedding, -> bf16
  k_scatter_add<<<NN, DGNN, 0, stream>>>(h_gnn_in, node_idx, sums, counts);
  k_select_subfeats<<<NN, DGNN, 0, stream>>>(sums, counts, node_emb, subf_bf);

  // 4) GEMM2: Wh (both heads) = sub_feats @ W_gat -> bf16 transposed WhT (256 x 4096)
  k_gemm_bf16<1><<<(NN / 16) * (DGNN / 64) / 4, 128, 0, stream>>>(
      subf_bf, WgatT, WhT, nullptr, NN, DGNN, DGNN);

  // 5) attention coefficients + safe-softmax stats
  k_attn_coef<<<(NHEAD * NN) / 256, 256, 0, stream>>>(WhT, a_gat, f_src, f_dst);
  k_gat_softmax_stats<<<(NHEAD * NN) / 8, 256, 0, stream>>>(adj, f_src, f_dst, rmax, rsum);

  // 6) WMMA attention aggregation, j-split 8 ways for occupancy (4096 waves)
  k_gat_aggregate<<<(NHEAD * (NN / 16) * JSPLIT) / 4, 128, 0, stream>>>(
      adj, f_src, f_dst, rmax, rsum, WhT, hp_part);

  // 7) chunk-reduce + ELU + LN -> sub_out ; gather + concat + LN2 -> z_bf16
  k_elu_ln<<<NN, DGNN, 0, stream>>>(hp_part, ln_g, ln_b, sub_out);
  k_fuse_ln<<<NN, 256, 0, stream>>>(h_temporal, sub_out, node_idx, ln2_g, ln2_b, z_bf);

  // 8) GEMM3 + GELU: z2 = gelu(z @ W1 + b1)   (4096x768x384)
  k_gemm_bf16<2><<<(NN / 16) * (FHID / 64) / 4, 128, 0, stream>>>(
      z_bf, W1T, z2, b1, NN, FHID, FUS);

  // 9) final head: out = z2 @ W2 + b2
  k_head<<<NN / 8, 256, 0, stream>>>(z2, W2, b2, out);
}
